// SABlock_15393162789516
// MI455X (gfx1250) — compile-verified, matrix operand order checked by reading, not run on hardware
//
#include <hip/hip_runtime.h>
#include <hip/hip_bf16.h>

// ---------------------------------------------------------------------------
// MI455X (gfx1250) transformer block:  QKV proj -> flash attention -> FFN ->
// residual.  All matmuls via v_wmma_f32_16x16x32_f16 (f16 in, f32 accum).
// GEMMs use a double-buffered LDS pipeline fed by GLOBAL_LOAD_ASYNC_TO_LDS
// (ASYNCcnt) when the toolchain exposes the builtin.
// Fragment layouts per CDNA5 ISA 7.12.2:
//   A (16x32 f16): lane<16 row=lane, K = {klo..klo+7, klo+16..klo+23}, klo=0;
//                  lane>=16 same row, klo=8.  (two b128 loads per lane)
//   B assumed mirror of A with lane index = N column.
//   C/D (16x16 f32): vgpr r, lanes 0-15 -> (M=r, N=lane); lanes 16-31 -> M=r+8.
// ---------------------------------------------------------------------------

typedef _Float16 v16h __attribute__((ext_vector_type(16)));
typedef float    v8f  __attribute__((ext_vector_type(8)));
typedef int      v4i  __attribute__((vector_size(16)));  // matches builtin proto

union HV { v16h v; uint4 u[2]; };
union H8 { _Float16 h[8]; uint4 u; };

__device__ __forceinline__ v8f wmma32f16(v16h a, v16h b, v8f c) {
  // (neg_a, A, neg_b, B, c_mod, C, reuse_a, reuse_b)
  return __builtin_amdgcn_wmma_f32_16x16x32_f16(false, a, false, b, (short)0, c,
                                                false, false);
}

// ---------------------------------------------------------------------------
// Async global->LDS 16-byte copy (CDNA5 GLOBAL_LOAD_ASYNC_TO_LDS_B128,
// tracked by ASYNCcnt).  Builtin proto (from hipcc diagnostic):
//   (v4i addrspace(1)* gsrc, v4i addrspace(3)* ldst, imm offset, imm cpol).
// AS3 pointer = low 32 bits of generic LDS address (ISA 10.2); AS1 pointer =
// generic global address.  Falls back to a synchronous b128 load+store
// (tracked by dscnt, which the barrier waits on) when unavailable.
// ---------------------------------------------------------------------------
#if __has_builtin(__builtin_amdgcn_global_load_async_to_lds_b128) && \
    __has_builtin(__builtin_amdgcn_s_wait_asynccnt)
#define HAVE_ASYNC_LDS 1
#else
#define HAVE_ASYNC_LDS 0
#endif

__device__ __forceinline__ void copy16_g2l(void* lds, const void* g) {
#if HAVE_ASYNC_LDS
  __builtin_amdgcn_global_load_async_to_lds_b128(
      (__attribute__((address_space(1))) v4i*)(uintptr_t)g,
      (__attribute__((address_space(3))) v4i*)(uint32_t)(uintptr_t)lds,
      0, 0);
#else
  *reinterpret_cast<uint4*>(lds) = *reinterpret_cast<const uint4*>(g);
#endif
}

__device__ __forceinline__ void wait_async_copies() {
#if HAVE_ASYNC_LDS
  __builtin_amdgcn_s_wait_asynccnt(0);
#endif
}

// ---------------------------------------------------------------------------
// f32 -> f16 conversion, vectorized x4 (all tensor sizes divisible by 4)
// ---------------------------------------------------------------------------
__global__ void cvt_f32_to_f16(const float* __restrict__ src,
                               _Float16* __restrict__ dst, int n4) {
  int stride = gridDim.x * blockDim.x;
  for (int i = blockIdx.x * blockDim.x + threadIdx.x; i < n4; i += stride) {
    float4 f = reinterpret_cast<const float4*>(src)[i];
    union { _Float16 h[4]; uint2 u; } p;
    p.h[0] = (_Float16)f.x; p.h[1] = (_Float16)f.y;
    p.h[2] = (_Float16)f.z; p.h[3] = (_Float16)f.w;
    reinterpret_cast<uint2*>(dst)[i] = p.u;
  }
}

// ---------------------------------------------------------------------------
// C[M,N] = act(A[M,K] @ W[N,K]^T + bias).  W row n is contiguous over K, so
// it is directly the WMMA B operand.  Block tile 128(M) x 128(N), 8 waves
// (2x4), each wave 64x32 = 4x2 WMMA tiles; K stepped by 32 through a
// double-buffered LDS stage filled by async copies (one barrier per step).
// OUT_MODE: 0 = f16 row-major, 1 = f16 transposed per head (V->Vt[b][h][d][s]),
//           2 = f32 + residual add (final output).
// ---------------------------------------------------------------------------
template <int OUT_MODE, bool RELU>
__global__ __launch_bounds__(256) void gemm_xwT(
    const _Float16* __restrict__ A, const _Float16* __restrict__ W,
    const float* __restrict__ bias, const float* __restrict__ resid,
    _Float16* __restrict__ outH, float* __restrict__ outF, int M, int N,
    int K) {
  __shared__ alignas(16) _Float16 sA[2][128][32];  // 2 x 8 KB
  __shared__ alignas(16) _Float16 sB[2][128][32];  // 2 x 8 KB

  const int tid = threadIdx.x;
  const int lane = tid & 31;
  const int w = tid >> 5;
  const int wm = w >> 2;             // 0..1  (64 rows per wave)
  const int wn = w & 3;              // 0..3  (32 cols per wave)
  const int lr = lane & 15;
  const int klo = (lane >> 4) << 3;  // 0 or 8
  const int m0 = blockIdx.y * 128;
  const int n0 = blockIdx.x * 128;

  // Per-thread copy slots: tiles are 128x32 halves = 512 b128 chunks each;
  // 256 threads -> 2 chunks of A + 2 chunks of B per thread per K-step.
  const int cr = tid >> 2;           // row 0..63 (+64 on second pass)
  const int cc = (tid & 3) << 3;     // half-offset 0/8/16/24

  v8f acc[4][2] = {};

  const int nk = K >> 5;
  // prefetch first K-slice into buffer 0
  {
    const size_t ga = (size_t)(m0 + cr) * K + cc;
    const size_t gb = (size_t)(n0 + cr) * K + cc;
    copy16_g2l(&sA[0][cr][cc], &A[ga]);
    copy16_g2l(&sA[0][cr + 64][cc], &A[ga + (size_t)64 * K]);
    copy16_g2l(&sB[0][cr][cc], &W[gb]);
    copy16_g2l(&sB[0][cr + 64][cc], &W[gb + (size_t)64 * K]);
  }

  for (int i = 0; i < nk; ++i) {
    const int cur = i & 1;
    wait_async_copies();   // my async writes into buffer `cur` are done
    __syncthreads();       // everyone's writes done + prev buffer fully read

    if (i + 1 < nk) {      // overlap next tile's global->LDS with the WMMAs
      const int k0 = (i + 1) << 5;
      const size_t ga = (size_t)(m0 + cr) * K + k0 + cc;
      const size_t gb = (size_t)(n0 + cr) * K + k0 + cc;
      copy16_g2l(&sA[cur ^ 1][cr][cc], &A[ga]);
      copy16_g2l(&sA[cur ^ 1][cr + 64][cc], &A[ga + (size_t)64 * K]);
      copy16_g2l(&sB[cur ^ 1][cr][cc], &W[gb]);
      copy16_g2l(&sB[cur ^ 1][cr + 64][cc], &W[gb + (size_t)64 * K]);
    }

    HV a[4], bf[2];
#pragma unroll
    for (int s = 0; s < 4; ++s) {
      const _Float16* pa = &sA[cur][wm * 64 + s * 16 + lr][0];
      a[s].u[0] = *reinterpret_cast<const uint4*>(pa + klo);
      a[s].u[1] = *reinterpret_cast<const uint4*>(pa + klo + 16);
    }
#pragma unroll
    for (int s = 0; s < 2; ++s) {
      const _Float16* pb = &sB[cur][wn * 32 + s * 16 + lr][0];
      bf[s].u[0] = *reinterpret_cast<const uint4*>(pb + klo);
      bf[s].u[1] = *reinterpret_cast<const uint4*>(pb + klo + 16);
    }
#pragma unroll
    for (int sm = 0; sm < 4; ++sm)
#pragma unroll
      for (int sn = 0; sn < 2; ++sn)
        acc[sm][sn] = wmma32f16(a[sm].v, bf[sn].v, acc[sm][sn]);
  }

#pragma unroll
  for (int sm = 0; sm < 4; ++sm) {
#pragma unroll
    for (int sn = 0; sn < 2; ++sn) {
      int nG = n0 + wn * 32 + sn * 16 + lr;   // column (fixed per lane)
      int mB = m0 + wm * 64 + sm * 16 + klo;  // first row of the 8 regs
      float bvv = bias ? bias[nG] : 0.0f;
      v8f c = acc[sm][sn];
      if (OUT_MODE == 1) {
        // Vt[b][h][d][s]: rows mB..mB+7 are contiguous s -> one b128 store.
        int hh = nG >> 6, dd = nG & 63;
        int bb = mB >> 11, ss = mB & 2047;
        H8 pk;
#pragma unroll
        for (int r = 0; r < 8; ++r) {
          float v = c[r] + bvv;
          if (RELU) v = fmaxf(v, 0.0f);
          pk.h[r] = (_Float16)v;
        }
        size_t off = ((((size_t)bb * 16 + hh) * 64 + dd) << 11) + ss;
        *reinterpret_cast<uint4*>(&outH[off]) = pk.u;
      } else if (OUT_MODE == 0) {
#pragma unroll
        for (int r = 0; r < 8; ++r) {
          float v = c[r] + bvv;
          if (RELU) v = fmaxf(v, 0.0f);
          outH[(size_t)(mB + r) * N + nG] = (_Float16)v;
        }
      } else {
#pragma unroll
        for (int r = 0; r < 8; ++r) {
          size_t off = (size_t)(mB + r) * N + nG;
          outF[off] = c[r] + bvv + resid[off];
        }
      }
    }
  }
}

// ---------------------------------------------------------------------------
// Flash attention: 1 wave per (b, h, 16-query tile).  32 keys/iteration
// (two 16x16 score tiles -> one 16x32 P fragment), online softmax, O
// accumulated as 16x64 f32.  V read from the pre-transposed Vt layout.
// ---------------------------------------------------------------------------
__global__ __launch_bounds__(32) void flash_attn(
    const _Float16* __restrict__ Q, const _Float16* __restrict__ Km,
    const _Float16* __restrict__ Vt, _Float16* __restrict__ AO) {
  constexpr int S = 2048, Dm = 1024;
  __shared__ alignas(16) _Float16 sP[16][32];  // P repack buffer (1 KB)

  const int lane = threadIdx.x;
  const int lr = lane & 15;
  const int klo = (lane >> 4) << 3;  // 0 or 8
  const int qt = blockIdx.x & 127;
  const int h = (blockIdx.x >> 7) & 15;
  const int b = blockIdx.x >> 11;
  const int qb = qt << 4;

  // Q fragments: A operand, rows = queries, K = head features (two 32-chunks)
  const _Float16* qrow = Q + ((size_t)b * S + qb + lr) * Dm + h * 64;
  HV qf0, qf1;
  qf0.u[0] = *reinterpret_cast<const uint4*>(qrow + klo);
  qf0.u[1] = *reinterpret_cast<const uint4*>(qrow + klo + 16);
  qf1.u[0] = *reinterpret_cast<const uint4*>(qrow + 32 + klo);
  qf1.u[1] = *reinterpret_cast<const uint4*>(qrow + 32 + klo + 16);

  float mrow[8], lrow[8], alpha[8];
  v8f O[4] = {};
#pragma unroll
  for (int r = 0; r < 8; ++r) { mrow[r] = -1e30f; lrow[r] = 0.0f; }

  const _Float16* kbase = Km + (size_t)b * S * Dm + h * 64;
  const _Float16* vbase = Vt + ((size_t)b * 16 + h) * 64 * S;

  for (int kb = 0; kb < S; kb += 32) {
    v8f s0 = {}, s1 = {};
    {
      const _Float16* kr = kbase + (size_t)(kb + lr) * Dm;
      HV k0, k1;
      k0.u[0] = *reinterpret_cast<const uint4*>(kr + klo);
      k0.u[1] = *reinterpret_cast<const uint4*>(kr + klo + 16);
      k1.u[0] = *reinterpret_cast<const uint4*>(kr + 32 + klo);
      k1.u[1] = *reinterpret_cast<const uint4*>(kr + 32 + klo + 16);
      s0 = wmma32f16(qf0.v, k0.v, s0);
      s0 = wmma32f16(qf1.v, k1.v, s0);
    }
    {
      const _Float16* kr = kbase + (size_t)(kb + 16 + lr) * Dm;
      HV k0, k1;
      k0.u[0] = *reinterpret_cast<const uint4*>(kr + klo);
      k0.u[1] = *reinterpret_cast<const uint4*>(kr + klo + 16);
      k1.u[0] = *reinterpret_cast<const uint4*>(kr + 32 + klo);
      k1.u[1] = *reinterpret_cast<const uint4*>(kr + 32 + klo + 16);
      s1 = wmma32f16(qf0.v, k0.v, s1);
      s1 = wmma32f16(qf1.v, k1.v, s1);
    }

    // Online softmax.  Row m of a score tile lives in one vgpr across a
    // 16-lane half-wave -> reduce with xor shuffles 1,2,4,8.
#pragma unroll
    for (int r = 0; r < 8; ++r) {
      float x0 = s0[r] * 0.125f;  // 1/sqrt(64)
      float x1 = s1[r] * 0.125f;
      float t = fmaxf(x0, x1);
      t = fmaxf(t, __shfl_xor(t, 1));
      t = fmaxf(t, __shfl_xor(t, 2));
      t = fmaxf(t, __shfl_xor(t, 4));
      t = fmaxf(t, __shfl_xor(t, 8));
      float mn = fmaxf(mrow[r], t);
      float a = __expf(mrow[r] - mn);
      x0 = __expf(x0 - mn);
      x1 = __expf(x1 - mn);
      float sm = x0 + x1;
      sm += __shfl_xor(sm, 1);
      sm += __shfl_xor(sm, 2);
      sm += __shfl_xor(sm, 4);
      sm += __shfl_xor(sm, 8);
      mrow[r] = mn;
      lrow[r] = lrow[r] * a + sm;
      alpha[r] = a;
      s0[r] = x0;
      s1[r] = x1;
    }

    // Repack P from C layout to A-operand layout through LDS.
    __syncthreads();
#pragma unroll
    for (int r = 0; r < 8; ++r) {
      sP[r + klo][lr] = (_Float16)s0[r];
      sP[r + klo][16 + lr] = (_Float16)s1[r];
    }
    __syncthreads();
    HV pf;
    pf.u[0] = *reinterpret_cast<const uint4*>(&sP[lr][klo]);
    pf.u[1] = *reinterpret_cast<const uint4*>(&sP[lr][klo + 16]);
    __syncthreads();

    // O[f] = alpha*O[f] + P @ V  (V columns from transposed layout, keys
    // contiguous -> straight b128 loads)
#pragma unroll
    for (int f = 0; f < 4; ++f) {
      const _Float16* vr = vbase + (size_t)(f * 16 + lr) * S + kb;
      HV vf;
      vf.u[0] = *reinterpret_cast<const uint4*>(vr + klo);
      vf.u[1] = *reinterpret_cast<const uint4*>(vr + klo + 16);
      v8f o = O[f];
#pragma unroll
      for (int r = 0; r < 8; ++r) o[r] *= alpha[r];
      O[f] = wmma32f16(pf.v, vf.v, o);
    }
  }

#pragma unroll
  for (int f = 0; f < 4; ++f) {
#pragma unroll
    for (int r = 0; r < 8; ++r) {
      float v = O[f][r] / lrow[r];
      AO[((size_t)b * S + qb + r + klo) * Dm + h * 64 + f * 16 + lr] =
          (_Float16)v;
    }
  }
}

// ---------------------------------------------------------------------------
// Host-side launcher.  Workspace layout (70 MB total):
//   Xh 8MB | Wqh 2 | Wkh 2 | Wvh 2 | W1h 4 | W2h 4 | Qh 8 | Kh 8 | Vt 8 |
//   AO 8 | Hh 16
// ---------------------------------------------------------------------------
extern "C" void kernel_launch(void* const* d_in, const int* in_sizes, int n_in,
                              void* d_out, int out_size, void* d_ws,
                              size_t ws_size, hipStream_t stream) {
  (void)in_sizes; (void)n_in; (void)out_size; (void)ws_size;
  const float* embed = (const float*)d_in[0];
  const float* Wq = (const float*)d_in[1];
  const float* Wk = (const float*)d_in[2];
  const float* Wv = (const float*)d_in[3];
  const float* bv = (const float*)d_in[4];
  const float* W1 = (const float*)d_in[5];
  const float* b1 = (const float*)d_in[6];
  const float* W2 = (const float*)d_in[7];
  const float* b2 = (const float*)d_in[8];

  constexpr size_t MB = 1ull << 20;
  char* ws = (char*)d_ws;
  _Float16* Xh  = (_Float16*)(ws + 0 * MB);
  _Float16* Wqh = (_Float16*)(ws + 8 * MB);
  _Float16* Wkh = (_Float16*)(ws + 10 * MB);
  _Float16* Wvh = (_Float16*)(ws + 12 * MB);
  _Float16* W1h = (_Float16*)(ws + 14 * MB);
  _Float16* W2h = (_Float16*)(ws + 18 * MB);
  _Float16* Qh  = (_Float16*)(ws + 22 * MB);
  _Float16* Kh  = (_Float16*)(ws + 30 * MB);
  _Float16* Vt  = (_Float16*)(ws + 38 * MB);
  _Float16* AO  = (_Float16*)(ws + 46 * MB);
  _Float16* Hh  = (_Float16*)(ws + 54 * MB);

  auto cvt = [&](const float* s, _Float16* d, int n) {
    int n4 = n >> 2;
    int blocks = (n4 + 255) / 256;
    if (blocks > 2048) blocks = 2048;
    cvt_f32_to_f16<<<dim3(blocks), dim3(256), 0, stream>>>(s, d, n4);
  };
  cvt(embed, Xh, 4096 * 1024);
  cvt(Wq, Wqh, 1024 * 1024);
  cvt(Wk, Wkh, 1024 * 1024);
  cvt(Wv, Wvh, 1024 * 1024);
  cvt(W1, W1h, 2097152);
  cvt(W2, W2h, 2097152);

  dim3 blk(256);
  // Q = X Wq^T ; K = X Wk^T ; Vt = transpose_head(X Wv^T + bv)
  gemm_xwT<0, false><<<dim3(8, 32), blk, 0, stream>>>(
      Xh, Wqh, nullptr, nullptr, Qh, nullptr, 4096, 1024, 1024);
  gemm_xwT<0, false><<<dim3(8, 32), blk, 0, stream>>>(
      Xh, Wkh, nullptr, nullptr, Kh, nullptr, 4096, 1024, 1024);
  gemm_xwT<1, false><<<dim3(8, 32), blk, 0, stream>>>(
      Xh, Wvh, bv, nullptr, Vt, nullptr, 4096, 1024, 1024);
  // attention: B*H*(S/16) = 4096 single-wave workgroups
  flash_attn<<<dim3(4096), dim3(32), 0, stream>>>(Qh, Kh, Vt, AO);
  // FFN: Hh = relu(AO W1^T + b1) ; out = Hh W2^T + b2 + embed
  gemm_xwT<0, true><<<dim3(16, 32), blk, 0, stream>>>(
      AO, W1h, b1, nullptr, Hh, nullptr, 4096, 2048, 1024);
  gemm_xwT<2, false><<<dim3(8, 32), blk, 0, stream>>>(
      Hh, W2h, b2, embed, nullptr, (float*)d_out, 4096, 1024, 2048);
}
